// SlotAttention_55989193671481
// MI455X (gfx1250) — matrix operand order repaired and checked by reading, not compiled, not run on hardware
//
#include <hip/hip_runtime.h>
#include <hip/hip_bf16.h>

#define DIM    256
#define NSEQ   4096
#define NB     64
#define NSLOT  8
#define HID    512
#define ITERS  3
#define EPSA   1e-8f
#define LN_EPS 1e-5f
#define SCALE  0.0625f   // 256^-0.5

typedef __bf16 bf16;
typedef __attribute__((ext_vector_type(16))) __bf16 v16bf;
typedef __attribute__((ext_vector_type(8)))  float  v8f;

union Frag {
    v16bf v;
    uint4 q[2];
};

__device__ inline float bf2f(bf16 x) { return (float)x; }
__device__ inline bf16  f2bf(float x) { return (bf16)x; }

__device__ inline float wave_sum(float v) {
#pragma unroll
    for (int m = 16; m >= 1; m >>= 1) v += __shfl_xor(v, m, 32);
    return v;
}

// -------------------------------------------------------------------------
// Fused LayerNorm(x) -> bf16 tile in LDS -> k = xln@wk^T+bk, v = xln@wv^T+bv
// Block: 256 threads (8 waves). M-tile = 128 rows. K = 256 (full row).
// -------------------------------------------------------------------------
__global__ __launch_bounds__(256) void lnkv_kernel(
    const float* __restrict__ x,
    const float* __restrict__ lng, const float* __restrict__ lnb,
    const bf16* __restrict__ wkb, const float* __restrict__ bk,
    const bf16* __restrict__ wvb, const float* __restrict__ bv,
    bf16* __restrict__ kout, bf16* __restrict__ vout)
{
    __shared__ __align__(16) bf16 xt[128 * DIM];   // 64 KB
    const int tid  = threadIdx.x;
    const int lane = tid & 31;
    const int wid  = tid >> 5;
    const long rowBase = (long)blockIdx.x * 128;

    // ---- Phase 1: LayerNorm 16 rows per wave, write bf16 to LDS ----
    const int c0 = lane * 8;
    float g[8], bb[8];
#pragma unroll
    for (int j = 0; j < 8; ++j) { g[j] = lng[c0 + j]; bb[j] = lnb[c0 + j]; }

    for (int r = 0; r < 16; ++r) {
        const int lr = wid * 16 + r;
        const float* xr = x + (rowBase + lr) * DIM + c0;
        float a[8];
        float s = 0.f;
#pragma unroll
        for (int j = 0; j < 8; ++j) { a[j] = xr[j]; s += a[j]; }
        const float mean = wave_sum(s) * (1.0f / DIM);
        float sq = 0.f;
#pragma unroll
        for (int j = 0; j < 8; ++j) { const float d = a[j] - mean; sq += d * d; }
        const float var = wave_sum(sq) * (1.0f / DIM);
        const float rs  = rsqrtf(var + LN_EPS);
        bf16* dst = xt + lr * DIM + c0;
#pragma unroll
        for (int j = 0; j < 8; ++j)
            dst[j] = f2bf((a[j] - mean) * rs * g[j] + bb[j]);
    }
    __syncthreads();

    // ---- Phase 2: WMMA GEMM. Wave strip = rows [wid*16, wid*16+16) ----
    const int m   = lane & 15;
    const int hi  = lane >> 4;
    const int kb  = hi * 8;    // A: K sub-chunk base (elements)
    const int kb2 = hi * 16;   // unused; B also uses kb
    const int arow = wid * 16 + m;

    Frag A[8];
#pragma unroll
    for (int c = 0; c < 8; ++c) {
        A[c].q[0] = *(const uint4*)(xt + arow * DIM + c * 32 + kb);
        A[c].q[1] = *(const uint4*)(xt + arow * DIM + c * 32 + kb + 16);
    }

    for (int nt = 0; nt < 16; ++nt) {
        const int n = nt * 16 + m;
        v8f ak = {};
        v8f av = {};
#pragma unroll
        for (int c = 0; c < 8; ++c) {
            Frag Bk, Bv;
            const uint4* pk = (const uint4*)(wkb + n * DIM + c * 32 + kb);
            Bk.q[0] = pk[0]; Bk.q[1] = pk[2];
            const uint4* pv = (const uint4*)(wvb + n * DIM + c * 32 + kb);
            Bv.q[0] = pv[0]; Bv.q[1] = pv[2];
            ak = __builtin_amdgcn_wmma_f32_16x16x32_bf16(false, A[c].v, false, Bk.v,
                                                         (short)0, ak, false, false);
            av = __builtin_amdgcn_wmma_f32_16x16x32_bf16(false, A[c].v, false, Bv.v,
                                                         (short)0, av, false, false);
        }
        const float bbk = bk[n];
        const float bbv = bv[n];
#pragma unroll
        for (int r = 0; r < 8; ++r) {
            const long row = rowBase + wid * 16 + r + 8 * hi;
            kout[row * DIM + n] = f2bf(ak[r] + bbk);
            vout[row * DIM + n] = f2bf(av[r] + bbv);
        }
    }
}

// -------------------------------------------------------------------------
// dots^T tile per wave: 16 n-rows x 16 slot-cols (8 valid), K=256 via WMMA.
// Then softmax over slots (16-lane butterfly), +EPS, store p, atomic rowsum.
// Grid: (32, 64) = (n-tiles of 128, batch). Block 256 = 8 waves x 16 n.
// -------------------------------------------------------------------------
__global__ __launch_bounds__(256) void attn_kernel(
    const bf16* __restrict__ kbf, const bf16* __restrict__ qbf,
    float* __restrict__ pbuf, float* __restrict__ rowsum)
{
    const int b    = blockIdx.y;
    const int lane = threadIdx.x & 31;
    const int wid  = threadIdx.x >> 5;
    const int nbase = blockIdx.x * 128 + wid * 16;
    const int m   = lane & 15;
    const int hi  = lane >> 4;
    const int kb  = hi * 8;
    const int kb2 = hi * 16;

    const bf16* krow = kbf + ((long)(b * NSEQ + nbase + m)) * DIM;
    const bf16* qrow = qbf + (long)(b * 16 + m) * DIM;

    v8f acc = {};
#pragma unroll
    for (int c = 0; c < 8; ++c) {
        Frag A, B;
        A.q[0] = *(const uint4*)(krow + c * 32 + kb);
        A.q[1] = *(const uint4*)(krow + c * 32 + kb + 16);
        const uint4* pq = (const uint4*)(qrow + c * 32 + kb);
        B.q[0] = pq[0]; B.q[1] = pq[2];
        acc = __builtin_amdgcn_wmma_f32_16x16x32_bf16(false, A.v, false, B.v,
                                                      (short)0, acc, false, false);
    }

    const int slot = m;           // column of dots^T tile
    float pv[8];
    float local = 0.f;
#pragma unroll
    for (int r = 0; r < 8; ++r) {
        float v = acc[r] * SCALE;
        if (slot >= NSLOT) v = -__builtin_inff();   // mask pad slots
        float mx = v;
#pragma unroll
        for (int msk = 8; msk >= 1; msk >>= 1)
            mx = fmaxf(mx, __shfl_xor(mx, msk, 32));
        const float e = __expf(v - mx);
        float s = e;
#pragma unroll
        for (int msk = 8; msk >= 1; msk >>= 1)
            s += __shfl_xor(s, msk, 32);
        const float pr = e / s + EPSA;
        pv[r] = pr;
        local += pr;
    }
    // combine the two n-halves (lane ^ 16 has the same slot)
    local += __shfl_xor(local, 16, 32);
    if (slot < NSLOT) {
        const int n0 = nbase + 8 * hi;
        float* pd = pbuf + ((long)(b * NSLOT + slot)) * NSEQ + n0;
#pragma unroll
        for (int r = 0; r < 8; ++r) pd[r] = pv[r];
        if (hi == 0) atomicAdd(&rowsum[b * NSLOT + slot], local);
    }
}

// -------------------------------------------------------------------------
// updates[b,s,d] += sum_n p[s,n] * v[n,d] over an n-slice of 512.
// Grid (64, 8). Thread owns one d. p staged through LDS (broadcast reads).
// -------------------------------------------------------------------------
__global__ __launch_bounds__(256) void upd_kernel(
    const bf16* __restrict__ vbf, const float* __restrict__ pbuf,
    float* __restrict__ updates)
{
    __shared__ float pb[NSLOT][64];
    const int b = blockIdx.x;
    const int n0 = blockIdx.y * 512;
    const int t = threadIdx.x;
    float acc[NSLOT];
#pragma unroll
    for (int s = 0; s < NSLOT; ++s) acc[s] = 0.f;

    for (int nc = 0; nc < 512; nc += 64) {
        __syncthreads();
        for (int i = t; i < NSLOT * 64; i += 256) {
            const int s = i >> 6, nn = i & 63;
            pb[s][nn] = pbuf[((long)(b * NSLOT + s)) * NSEQ + n0 + nc + nn];
        }
        __syncthreads();
        const bf16* vp = vbf + ((long)(b * NSEQ + n0 + nc)) * DIM + t;
        for (int nn = 0; nn < 64; ++nn) {
            const float vvv = bf2f(vp[(long)nn * DIM]);
#pragma unroll
            for (int s = 0; s < NSLOT; ++s) acc[s] += pb[s][nn] * vvv;
        }
    }
#pragma unroll
    for (int s = 0; s < NSLOT; ++s)
        atomicAdd(&updates[(b * NSLOT + s) * DIM + t], acc[s]);
}

// -------------------------------------------------------------------------
// q = LN(slots) @ wq^T + bq -> bf16, padded to 16 slot rows (zeros).
// Grid 64 (batch). 8 waves: wave w LayerNorms slot row w; then each thread
// computes one output column j for all 8 slots.
// -------------------------------------------------------------------------
__global__ __launch_bounds__(256) void q_kernel(
    const float* __restrict__ slots,
    const float* __restrict__ lng, const float* __restrict__ lnb,
    const float* __restrict__ wq, const float* __restrict__ bq,
    bf16* __restrict__ qbf)
{
    __shared__ float lnS[NSLOT][DIM];
    const int b    = blockIdx.x;
    const int lane = threadIdx.x & 31;
    const int wid  = threadIdx.x >> 5;

    {
        const int c0 = lane * 8;
        const float* sr = slots + (b * NSLOT + wid) * DIM + c0;
        float a[8];
        float s = 0.f;
#pragma unroll
        for (int j = 0; j < 8; ++j) { a[j] = sr[j]; s += a[j]; }
        const float mean = wave_sum(s) * (1.0f / DIM);
        float sq = 0.f;
#pragma unroll
        for (int j = 0; j < 8; ++j) { const float d = a[j] - mean; sq += d * d; }
        const float var = wave_sum(sq) * (1.0f / DIM);
        const float rs  = rsqrtf(var + LN_EPS);
#pragma unroll
        for (int j = 0; j < 8; ++j)
            lnS[wid][c0 + j] = (a[j] - mean) * rs * lng[c0 + j] + lnb[c0 + j];
    }
    __syncthreads();

    const int t = threadIdx.x;          // output feature j
    float acc[NSLOT];
#pragma unroll
    for (int s = 0; s < NSLOT; ++s) acc[s] = 0.f;
    const float* wr = wq + (long)t * DIM;
    for (int d = 0; d < DIM; ++d) {
        const float wvv = wr[d];
#pragma unroll
        for (int s = 0; s < NSLOT; ++s) acc[s] += wvv * lnS[s][d];
    }
    const float bjq = bq[t];
#pragma unroll
    for (int s = 0; s < NSLOT; ++s)
        qbf[(long)(b * 16 + s) * DIM + t] = f2bf(acc[s] + bjq);
#pragma unroll
    for (int s = NSLOT; s < 16; ++s)
        qbf[(long)(b * 16 + s) * DIM + t] = f2bf(0.f);
}

// -------------------------------------------------------------------------
// Fused GRUCell + LN + MLP residual for one (batch, slot) row per block.
// -------------------------------------------------------------------------
__global__ __launch_bounds__(256) void gru_mlp_kernel(
    const float* __restrict__ updates, const float* __restrict__ rowsum,
    const float* __restrict__ wih, const float* __restrict__ bih,
    const float* __restrict__ whh, const float* __restrict__ bhh,
    const float* __restrict__ w1,  const float* __restrict__ b1,
    const float* __restrict__ w2,  const float* __restrict__ b2,
    const float* __restrict__ lng, const float* __restrict__ lnb,
    float* __restrict__ slots)
{
    __shared__ float xu[DIM], hp[DIM], gi[3 * DIM], gh[3 * DIM];
    __shared__ float hn[DIM], lnh[DIM], mh[HID];
    __shared__ float red[8];

    const int bs = blockIdx.x;       // 0..511 = (batch*8 + slot)
    const int t  = threadIdx.x;

    const float inv = 1.0f / rowsum[bs];
    xu[t] = updates[bs * DIM + t] * inv;
    hp[t] = slots[bs * DIM + t];
    __syncthreads();

    // gi = updates @ w_ih^T + b_ih ; gh = h @ w_hh^T + b_hh  (j = t, t+256, t+512)
#pragma unroll
    for (int jj = 0; jj < 3; ++jj) {
        const int j = jj * DIM + t;
        float a = bih[j], c = bhh[j];
        const float* wi = wih + (long)j * DIM;
        const float* wh = whh + (long)j * DIM;
        for (int d = 0; d < DIM; ++d) { a += wi[d] * xu[d]; c += wh[d] * hp[d]; }
        gi[j] = a; gh[j] = c;
    }
    __syncthreads();

    {
        const float r = 1.0f / (1.0f + __expf(-(gi[t] + gh[t])));
        const float z = 1.0f / (1.0f + __expf(-(gi[DIM + t] + gh[DIM + t])));
        const float n = tanhf(gi[2 * DIM + t] + r * gh[2 * DIM + t]);
        hn[t] = (1.0f - z) * n + z * hp[t];
    }
    __syncthreads();

    // LayerNorm over hn
    const float h = hn[t];
    float s = wave_sum(h);
    if ((t & 31) == 0) red[t >> 5] = s;
    __syncthreads();
    float mean = 0.f;
#pragma unroll
    for (int i = 0; i < 8; ++i) mean += red[i];
    mean *= (1.0f / DIM);
    __syncthreads();
    const float d0 = h - mean;
    float sq = wave_sum(d0 * d0);
    if ((t & 31) == 0) red[t >> 5] = sq;
    __syncthreads();
    float var = 0.f;
#pragma unroll
    for (int i = 0; i < 8; ++i) var += red[i];
    var *= (1.0f / DIM);
    const float rs = rsqrtf(var + LN_EPS);
    lnh[t] = d0 * rs * lng[t] + lnb[t];
    __syncthreads();

    // MLP layer 1 (512 outs, 2 per thread) + ReLU
#pragma unroll
    for (int jj = 0; jj < 2; ++jj) {
        const int j = jj * DIM + t;
        float a = b1[j];
        const float* wr = w1 + (long)j * DIM;
        for (int d = 0; d < DIM; ++d) a += wr[d] * lnh[d];
        mh[j] = fmaxf(a, 0.f);
    }
    __syncthreads();

    // MLP layer 2 + residual
    float o = b2[t];
    const float* wr2 = w2 + (long)t * HID;
    for (int j = 0; j < HID; ++j) o += wr2[j] * mh[j];
    slots[bs * DIM + t] = hn[t] + o;
}

// -------------------------------------------------------------------------
// Tiny helper kernels
// -------------------------------------------------------------------------
__global__ void init_slots_kernel(const float* __restrict__ si,
                                  const float* __restrict__ mu,
                                  const float* __restrict__ sg,
                                  float* __restrict__ slots)
{
    const int i = blockIdx.x * 256 + threadIdx.x;   // 131072 total
    const int d = i & (DIM - 1);
    slots[i] = mu[d] + sg[d] * si[i];
}

__global__ void conv_w_kernel(const float* __restrict__ wk,
                              const float* __restrict__ wv,
                              bf16* __restrict__ wkb, bf16* __restrict__ wvb)
{
    const int i = blockIdx.x * 256 + threadIdx.x;   // 65536 total
    wkb[i] = f2bf(wk[i]);
    wvb[i] = f2bf(wv[i]);
}

__global__ void zero_kernel(float* __restrict__ updates, float* __restrict__ rowsum)
{
    const int i = blockIdx.x * 256 + threadIdx.x;   // 131072 total
    if (i < NB * NSLOT) rowsum[i] = 0.f;
    updates[i] = 0.f;                                // NB*NSLOT*DIM == 131072
}

// -------------------------------------------------------------------------
extern "C" void kernel_launch(void* const* d_in, const int* in_sizes, int n_in,
                              void* d_out, int out_size, void* d_ws, size_t ws_size,
                              hipStream_t stream)
{
    const float* inputs     = (const float*)d_in[0];
    const float* slots_init = (const float*)d_in[1];
    const float* slots_mu   = (const float*)d_in[2];
    const float* slots_sig  = (const float*)d_in[3];
    const float* wq   = (const float*)d_in[4];
    const float* bq   = (const float*)d_in[5];
    const float* wk   = (const float*)d_in[6];
    const float* bk   = (const float*)d_in[7];
    const float* wv   = (const float*)d_in[8];
    const float* bv   = (const float*)d_in[9];
    const float* w_ih = (const float*)d_in[10];
    const float* b_ih = (const float*)d_in[11];
    const float* w_hh = (const float*)d_in[12];
    const float* b_hh = (const float*)d_in[13];
    const float* mlp_w1 = (const float*)d_in[14];
    const float* mlp_b1 = (const float*)d_in[15];
    const float* mlp_w2 = (const float*)d_in[16];
    const float* mlp_b2 = (const float*)d_in[17];
    const float* ln_in_g    = (const float*)d_in[18];
    const float* ln_in_b    = (const float*)d_in[19];
    const float* ln_slots_g = (const float*)d_in[20];
    const float* ln_slots_b = (const float*)d_in[21];
    const float* ln_ff_g    = (const float*)d_in[22];
    const float* ln_ff_b    = (const float*)d_in[23];

    char* ws = (char*)d_ws;
    size_t off = 0;
    auto alloc = [&](size_t bytes) -> void* {
        void* p = ws + off;
        off = (off + bytes + 255) & ~(size_t)255;
        return p;
    };

    bf16*  kbf     = (bf16*)alloc((size_t)NB * NSEQ * DIM * 2);   // 134 MB
    bf16*  vbf     = (bf16*)alloc((size_t)NB * NSEQ * DIM * 2);   // 134 MB
    bf16*  wkb     = (bf16*)alloc((size_t)DIM * DIM * 2);
    bf16*  wvb     = (bf16*)alloc((size_t)DIM * DIM * 2);
    bf16*  qbf     = (bf16*)alloc((size_t)NB * 16 * DIM * 2);
    float* pbuf    = (float*)alloc((size_t)NB * NSLOT * NSEQ * 4); // 8 MB
    float* rowsum  = (float*)alloc((size_t)NB * NSLOT * 4);
    float* updates = (float*)alloc((size_t)NB * NSLOT * DIM * 4);
    (void)ws_size; (void)in_sizes; (void)n_in; (void)out_size;

    float* slots = (float*)d_out;    // [64, 8, 256] lives in the output buffer

    conv_w_kernel<<<256, 256, 0, stream>>>(wk, wv, wkb, wvb);
    init_slots_kernel<<<512, 256, 0, stream>>>(slots_init, slots_mu, slots_sig, slots);
    lnkv_kernel<<<(NB * NSEQ) / 128, 256, 0, stream>>>(
        inputs, ln_in_g, ln_in_b, wkb, bk, wvb, bv, kbf, vbf);

    for (int it = 0; it < ITERS; ++it) {
        q_kernel<<<NB, 256, 0, stream>>>(slots, ln_slots_g, ln_slots_b, wq, bq, qbf);
        zero_kernel<<<512, 256, 0, stream>>>(updates, rowsum);
        attn_kernel<<<dim3(NSEQ / 128, NB), 256, 0, stream>>>(kbf, qbf, pbuf, rowsum);
        upd_kernel<<<dim3(NB, NSEQ / 512), 256, 0, stream>>>(vbf, pbuf, updates);
        gru_mlp_kernel<<<NB * NSLOT, 256, 0, stream>>>(
            updates, rowsum, w_ih, b_ih, w_hh, b_hh,
            mlp_w1, mlp_b1, mlp_w2, mlp_b2, ln_ff_g, ln_ff_b, slots);
    }
}
